// Block_87986700026383
// MI455X (gfx1250) — matrix operand order repaired and checked by reading, not compiled, not run on hardware
//
#include <hip/hip_runtime.h>
#include <hip/hip_bf16.h>
#include <math.h>

// ---------------- problem constants ----------------
#define BB    4
#define NSEQ  2048
#define DD    1024
#define HH    16
#define DHD   64
#define INNER 1024           // H*DH
#define FFD   4096
#define MM    (BB * NSEQ)    // 8192 rows

// ---------------- WMMA types ----------------
typedef __attribute__((ext_vector_type(16))) __bf16 v16bf;
typedef __attribute__((ext_vector_type(8)))  float  v8f;
typedef __attribute__((ext_vector_type(4)))  unsigned int u32x4;
typedef __attribute__((ext_vector_type(8)))  unsigned int u32x8;

union FragU { unsigned int u[8]; v16bf v; };

__device__ __forceinline__ unsigned short f2bf(float f) {
    unsigned int u = __float_as_uint(f);
    unsigned int r = u + 0x7FFFu + ((u >> 16) & 1u);   // round-to-nearest-even
    return (unsigned short)(r >> 16);
}

// A fragment (16x32 bf16, row-major source, ld = row stride in elements)
// lane m = lane&15 ; half = lane>>4 selects K subsets {0..7,16..23} / {8..15,24..31}
__device__ __forceinline__ v16bf load_frag_a(const unsigned short* A, int ld,
                                             size_t row, int half, int kb) {
    FragU r;
    const unsigned short* p0 = A + row * (size_t)ld + kb + 8 * half;
    const unsigned short* p1 = p0 + 16;
#pragma unroll
    for (int i = 0; i < 4; ++i) {
        r.u[i]     = *(const unsigned int*)(p0 + 2 * i);
        r.u[4 + i] = *(const unsigned int*)(p1 + 2 * i);
    }
    return r.v;
}

// B fragment (32x16 bf16) from n-major storage WT[n][k], ld = K stride.
// lane n = lane&15 ; half selects K subsets {0..7,16..23} / {8..15,24..31}
__device__ __forceinline__ v16bf load_frag_b(const unsigned short* W, int ld,
                                             size_t nrow, int half, int kb) {
    FragU r;
    const unsigned short* p = W + nrow * (size_t)ld + kb + 8 * half;
#pragma unroll
    for (int i = 0; i < 4; ++i) {
        r.u[i]     = *(const unsigned int*)(p + 2 * i);
        r.u[4 + i] = *(const unsigned int*)(p + 16 + 2 * i);
    }
    return r.v;
}

__device__ __forceinline__ v8f wmma_bf16(v16bf a, v16bf b, v8f c) {
    return __builtin_amdgcn_wmma_f32_16x16x32_bf16(false, a, false, b,
                                                   (short)0, c, false, false);
}

// ---- Tensor Data Mover: DMA a 32-row x 64-bf16 2D tile (row stride 64) to LDS ----
__device__ __forceinline__ void tdm_load_k_tile(const unsigned short* gsrc,
                                                unsigned short* lds_dst) {
    unsigned long long ga = (unsigned long long)(size_t)gsrc;
    unsigned int la = (unsigned int)(size_t)lds_dst;
    u32x4 g0;
    g0[0] = 1u;                                  // count=1 valid descriptor
    g0[1] = la;                                  // lds_addr (bytes)
    g0[2] = (unsigned int)ga;                    // global_addr[31:0]
    g0[3] = ((unsigned int)(ga >> 32) & 0x01FFFFFFu) | 0x80000000u; // addr[56:32] | type=2
    u32x8 g1;
    g1[0] = 0x00010000u;       // data_size=1 (2 bytes/elem)
    g1[1] = (64u << 16);       // tensor_dim0 = 64       (bits 79:48 low part)
    g1[2] = ((unsigned)NSEQ << 16); // tensor_dim1 = 2048 (bits 111:80 low part)
    g1[3] = (64u << 16);       // tile_dim0 = 64         (bits 127:112)
    g1[4] = 32u;               // tile_dim1 = 32, tile_dim2 = 0
    g1[5] = 64u;               // tensor_dim0_stride = 64
    g1[6] = 0u;
    g1[7] = 0u;
    asm volatile("tensor_load_to_lds %0, %1" :: "s"(g0), "s"(g1) : "memory");
}

// 32x64 tile GEMM core: two A-frags reuse each B-frag -> 8 WMMAs per k-step
__device__ __forceinline__ void gemm_core32(const unsigned short* A,
                                            const unsigned short* W, int K,
                                            int mbase, int nbase, int lane,
                                            v8f c[2][4]) {
    int l15 = lane & 15, half = lane >> 4;
    for (int kb = 0; kb < K; kb += 32) {
        int pk = kb + 64;                       // branch-free clamped prefetch
        if (pk > K - 32) pk = K - 32;
        __builtin_prefetch(A + (size_t)(mbase + l15) * K + pk, 0, 1);
        __builtin_prefetch(A + (size_t)(mbase + 16 + l15) * K + pk, 0, 1);
        v16bf va0 = load_frag_a(A, K, (size_t)(mbase + l15), half, kb);
        v16bf va1 = load_frag_a(A, K, (size_t)(mbase + 16 + l15), half, kb);
#pragma unroll
        for (int f = 0; f < 4; ++f) {
            v16bf vb = load_frag_b(W, K, (size_t)(nbase + f * 16 + l15), half, kb);
            c[0][f] = wmma_bf16(va0, vb, c[0][f]);
            c[1][f] = wmma_bf16(va1, vb, c[1][f]);
        }
    }
}

// ---------------- weight convert + transpose: f32 [K,N] -> bf16 [N,K] ----------------
__global__ __launch_bounds__(256) void wconv_kernel(const float* __restrict__ W,
                                                    unsigned short* __restrict__ WT,
                                                    int K, int Ncols) {
    size_t idx = (size_t)blockIdx.x * 256 + threadIdx.x;
    size_t total = (size_t)K * Ncols;
    if (idx >= total) return;
    size_t k = idx / Ncols, n = idx % Ncols;
    WT[n * (size_t)K + k] = f2bf(W[idx]);
}

// ---------------- LayerNorm: f32 row -> bf16 row ----------------
__global__ __launch_bounds__(256) void ln_kernel(const float* __restrict__ x,
                                                 const float* __restrict__ g,
                                                 const float* __restrict__ beta,
                                                 unsigned short* __restrict__ out) {
    __shared__ float s1[256], s2[256];
    int tid = threadIdx.x;
    size_t row = blockIdx.x;
    const float* xr = x + row * DD;
    float sum = 0.f, sq = 0.f;
    for (int i = tid; i < DD; i += 256) { float v = xr[i]; sum += v; sq += v * v; }
    s1[tid] = sum; s2[tid] = sq; __syncthreads();
    for (int off = 128; off; off >>= 1) {
        if (tid < off) { s1[tid] += s1[tid + off]; s2[tid] += s2[tid + off]; }
        __syncthreads();
    }
    float mu  = s1[0] * (1.0f / DD);
    float var = s2[0] * (1.0f / DD) - mu * mu;
    float inv = rsqrtf(var + 1e-5f);
    for (int i = tid; i < DD; i += 256)
        out[row * DD + i] = f2bf((xr[i] - mu) * inv * g[i] + beta[i]);
}

// ---------------- QKV GEMM, scatter into Q [B,H,N,DH], K [B,H,N,DH], V^T [B,H,DH,N] ----------------
__global__ __launch_bounds__(32) void gemm_qkv_kernel(const unsigned short* __restrict__ hb,
                                                      const unsigned short* __restrict__ wT,
                                                      unsigned short* __restrict__ qb,
                                                      unsigned short* __restrict__ kb,
                                                      unsigned short* __restrict__ vtb) {
    int lane = threadIdx.x, l15 = lane & 15, half = lane >> 4;
    int mbase = blockIdx.x * 32, nbase = blockIdx.y * 64;
    v8f c[2][4];
#pragma unroll
    for (int r = 0; r < 2; ++r)
#pragma unroll
        for (int f = 0; f < 4; ++f) { v8f z = {}; c[r][f] = z; }
    gemm_core32(hb, wT, DD, mbase, nbase, lane, c);
#pragma unroll
    for (int r = 0; r < 2; ++r)
#pragma unroll
        for (int f = 0; f < 4; ++f)
#pragma unroll
            for (int v = 0; v < 8; ++v) {
                int row = mbase + r * 16 + v + 8 * half;
                int b = row / NSEQ, n = row % NSEQ;
                int col = nbase + f * 16 + l15;
                unsigned short bv = f2bf(c[r][f][v]);
                if (col < INNER) {
                    int hd = col >> 6, d = col & 63;
                    qb[(((size_t)(b * HH + hd)) * NSEQ + n) * DHD + d] = bv;
                } else if (col < 2 * INNER) {
                    int cc = col - INNER, hd = cc >> 6, d = cc & 63;
                    kb[(((size_t)(b * HH + hd)) * NSEQ + n) * DHD + d] = bv;
                } else {
                    int cc = col - 2 * INNER, hd = cc >> 6, d = cc & 63;
                    vtb[(((size_t)(b * HH + hd)) * DHD + d) * NSEQ + n] = bv;
                }
            }
}

// ---------------- flash-style causal attention, one wave per 16-row tile ----------------
// K-stripes (32x64 bf16) are double-buffered into LDS by the Tensor Data Mover.
__global__ __launch_bounds__(32) void attn_kernel(const unsigned short* __restrict__ qb,
                                                  const unsigned short* __restrict__ kb,
                                                  const unsigned short* __restrict__ vtb,
                                                  unsigned short* __restrict__ attn_out) {
    __shared__ unsigned short k_lds[2 * 32 * DHD];   // 8 KB double buffer
    __shared__ unsigned short p_lds[16 * 32];        // 1 KB P-tile relayout
    int lane = threadIdx.x, l15 = lane & 15, half = lane >> 4;
    int tile = blockIdx.x;
    int it = tile % (NSEQ / 16);
    int bh = tile / (NSEQ / 16);
    int hh = bh % HH, b = bh / HH;
    int ibase = it * 16;
    const unsigned short* qh  = qb  + (size_t)(b * HH + hh) * NSEQ * DHD;
    const unsigned short* khd = kb  + (size_t)(b * HH + hh) * NSEQ * DHD;
    const unsigned short* vth = vtb + (size_t)(b * HH + hh) * DHD * NSEQ;

    const float scale = 0.125f; // DH^-0.5
    float mrun[8], lrun[8];
    v8f acc[4];
#pragma unroll
    for (int f = 0; f < 4; ++f) { v8f z = {}; acc[f] = z; }
#pragma unroll
    for (int v = 0; v < 8; ++v) { mrun[v] = -3.0e38f; lrun[v] = 0.f; }

    int nsteps = (ibase + 16 + 31) / 32;
    // prologue: DMA first K stripe
    tdm_load_k_tile(khd, k_lds);

    for (int step = 0; step < nsteps; ++step) {
        int jb = step * 32;
        int stage = step & 1;
        const unsigned short* kst = k_lds + stage * (32 * DHD);
        bool has_next = (step + 1 < nsteps);
        if (has_next) {   // DMA next stripe while computing this one
            tdm_load_k_tile(khd + (size_t)(jb + 32) * DHD,
                            k_lds + (stage ^ 1) * (32 * DHD));
            __builtin_amdgcn_s_wait_tensorcnt(1);
        } else {
            __builtin_amdgcn_s_wait_tensorcnt(0);
        }
        asm volatile("" ::: "memory");

        v8f c0 = {}, c1 = {};
        for (int ks = 0; ks < DHD; ks += 32) {
            v16bf aq = load_frag_a(qh, DHD, (size_t)(ibase + l15), half, ks);
            v16bf b0 = load_frag_b(kst, DHD, (size_t)l15, half, ks);
            v16bf b1 = load_frag_b(kst, DHD, (size_t)(16 + l15), half, ks);
            c0 = wmma_bf16(aq, b0, c0);
            c1 = wmma_bf16(aq, b1, c1);
        }
        bool need_mask = (jb + 32 > ibase);
#pragma unroll
        for (int v = 0; v < 8; ++v) {
            float s0 = c0[v] * scale, s1 = c1[v] * scale;
            if (need_mask) {
                int ig = ibase + v + 8 * half;
                if (jb + l15 > ig)      s0 = -3.0e38f;
                if (jb + 16 + l15 > ig) s1 = -3.0e38f;
            }
            c0[v] = s0; c1[v] = s1;
        }
        // row max across the 16-lane group
        float mt[8];
#pragma unroll
        for (int v = 0; v < 8; ++v) mt[v] = fmaxf(c0[v], c1[v]);
        for (int msk = 1; msk < 16; msk <<= 1)
#pragma unroll
            for (int v = 0; v < 8; ++v) mt[v] = fmaxf(mt[v], __shfl_xor(mt[v], msk, 32));
        float fac[8];
#pragma unroll
        for (int v = 0; v < 8; ++v) {
            float nm = fmaxf(mrun[v], mt[v]);
            fac[v]  = __expf(mrun[v] - nm);
            mrun[v] = nm;
            c0[v] = __expf(c0[v] - nm);
            c1[v] = __expf(c1[v] - nm);
        }
        float rs[8];
#pragma unroll
        for (int v = 0; v < 8; ++v) rs[v] = c0[v] + c1[v];
        for (int msk = 1; msk < 16; msk <<= 1)
#pragma unroll
            for (int v = 0; v < 8; ++v) rs[v] += __shfl_xor(rs[v], msk, 32);
#pragma unroll
        for (int v = 0; v < 8; ++v) lrun[v] = lrun[v] * fac[v] + rs[v];
#pragma unroll
        for (int f = 0; f < 4; ++f)
#pragma unroll
            for (int v = 0; v < 8; ++v) acc[f][v] *= fac[v];

        // P (16x32) -> bf16 via LDS to re-layout C-frag -> A-frag
        __syncthreads();
#pragma unroll
        for (int v = 0; v < 8; ++v) {
            int m = v + 8 * half;
            p_lds[m * 32 + l15]      = f2bf(c0[v]);
            p_lds[m * 32 + 16 + l15] = f2bf(c1[v]);
        }
        __syncthreads();
        v16bf ap = load_frag_a(p_lds, 32, (size_t)l15, half, 0);
#pragma unroll
        for (int f = 0; f < 4; ++f) {
            v16bf bv = load_frag_b(vth, NSEQ, (size_t)(f * 16 + l15), half, jb);
            acc[f] = wmma_bf16(ap, bv, acc[f]);
        }
    }
    // normalize + write bf16 attention output [M, INNER]
#pragma unroll
    for (int f = 0; f < 4; ++f)
#pragma unroll
        for (int v = 0; v < 8; ++v) {
            int m = v + 8 * half;
            size_t row = (size_t)b * NSEQ + ibase + m;
            attn_out[row * INNER + hh * 64 + f * 16 + l15] = f2bf(acc[f][v] / lrun[v]);
        }
}

// ---------------- out-proj GEMM + bias + residual -> f32 x1 ----------------
__global__ __launch_bounds__(32) void gemm_outproj_kernel(const unsigned short* __restrict__ ab,
                                                          const unsigned short* __restrict__ wT,
                                                          const float* __restrict__ bias,
                                                          const float* __restrict__ x0,
                                                          float* __restrict__ x1) {
    int lane = threadIdx.x, l15 = lane & 15, half = lane >> 4;
    int mbase = blockIdx.x * 32, nbase = blockIdx.y * 64;
    v8f c[2][4];
#pragma unroll
    for (int r = 0; r < 2; ++r)
#pragma unroll
        for (int f = 0; f < 4; ++f) { v8f z = {}; c[r][f] = z; }
    gemm_core32(ab, wT, INNER, mbase, nbase, lane, c);
#pragma unroll
    for (int r = 0; r < 2; ++r)
#pragma unroll
        for (int f = 0; f < 4; ++f)
#pragma unroll
            for (int v = 0; v < 8; ++v) {
                size_t row = (size_t)mbase + r * 16 + v + 8 * half;
                int col = nbase + f * 16 + l15;
                x1[row * DD + col] = c[r][f][v] + bias[col] + x0[row * DD + col];
            }
}

// ---------------- FF1 GEMM + bias + exact GELU -> bf16 ----------------
__global__ __launch_bounds__(32) void gemm_ff1_kernel(const unsigned short* __restrict__ hb,
                                                      const unsigned short* __restrict__ wT,
                                                      const float* __restrict__ bias,
                                                      unsigned short* __restrict__ g) {
    int lane = threadIdx.x, l15 = lane & 15, half = lane >> 4;
    int mbase = blockIdx.x * 32, nbase = blockIdx.y * 64;
    v8f c[2][4];
#pragma unroll
    for (int r = 0; r < 2; ++r)
#pragma unroll
        for (int f = 0; f < 4; ++f) { v8f z = {}; c[r][f] = z; }
    gemm_core32(hb, wT, DD, mbase, nbase, lane, c);
#pragma unroll
    for (int r = 0; r < 2; ++r)
#pragma unroll
        for (int f = 0; f < 4; ++f)
#pragma unroll
            for (int v = 0; v < 8; ++v) {
                size_t row = (size_t)mbase + r * 16 + v + 8 * half;
                int col = nbase + f * 16 + l15;
                float z = c[r][f][v] + bias[col];
                float gz = 0.5f * z * (1.0f + erff(z * 0.70710678118654752f));
                g[row * FFD + col] = f2bf(gz);
            }
}

// ---------------- FF2 GEMM + bias + residual -> f32 out ----------------
__global__ __launch_bounds__(32) void gemm_ff2_kernel(const unsigned short* __restrict__ g,
                                                      const unsigned short* __restrict__ wT,
                                                      const float* __restrict__ bias,
                                                      const float* __restrict__ x1,
                                                      float* __restrict__ xo) {
    int lane = threadIdx.x, l15 = lane & 15, half = lane >> 4;
    int mbase = blockIdx.x * 32, nbase = blockIdx.y * 64;
    v8f c[2][4];
#pragma unroll
    for (int r = 0; r < 2; ++r)
#pragma unroll
        for (int f = 0; f < 4; ++f) { v8f z = {}; c[r][f] = z; }
    gemm_core32(g, wT, FFD, mbase, nbase, lane, c);
#pragma unroll
    for (int r = 0; r < 2; ++r)
#pragma unroll
        for (int f = 0; f < 4; ++f)
#pragma unroll
            for (int v = 0; v < 8; ++v) {
                size_t row = (size_t)mbase + r * 16 + v + 8 * half;
                int col = nbase + f * 16 + l15;
                xo[row * DD + col] = c[r][f][v] + bias[col] + x1[row * DD + col];
            }
}

// ---------------- halt logits ----------------
__global__ __launch_bounds__(256) void halt_kernel(const float* __restrict__ x2,
                                                   const float* __restrict__ wh,
                                                   const float* __restrict__ bh,
                                                   float* __restrict__ halt) {
    __shared__ float sred[256];
    int tid = threadIdx.x, b = blockIdx.x;
    const float* xb = x2 + (size_t)b * NSEQ * DD;
    float acc = 0.f;
    for (size_t i = tid; i < (size_t)NSEQ * DD; i += 256)
        acc += xb[i] * wh[i & (DD - 1)];
    sred[tid] = acc; __syncthreads();
    for (int off = 128; off; off >>= 1) {
        if (tid < off) sred[tid] += sred[tid + off];
        __syncthreads();
    }
    if (tid == 0) halt[b] = sred[0] * (1.0f / NSEQ) + bh[0];
}

// ---------------- workspace layout (bytes) ----------------
static const size_t OFF_WQKVT = 0;                                   // 3072*1024*2 = 6291456
static const size_t OFF_WOUTT = 6291456;                             // 1024*1024*2 = 2097152
static const size_t OFF_WFF1T = 8388608;                             // 4096*1024*2 = 8388608
static const size_t OFF_WFF2T = 16777216;                            // 1024*4096*2 = 8388608
static const size_t OFF_A     = 25165824;                            // 16 MiB: h / attn_out / h2
static const size_t OFF_Q     = 41943040;                            // 16 MiB
static const size_t OFF_K     = 58720256;                            // 16 MiB
static const size_t OFF_V     = 75497472;                            // 16 MiB
static const size_t OFF_G     = 41943040;                            // 64 MiB, overlays dead Q/K/V
static const size_t OFF_X1    = 109051904;                           // 32 MiB f32

extern "C" void kernel_launch(void* const* d_in, const int* in_sizes, int n_in,
                              void* d_out, int out_size, void* d_ws, size_t ws_size,
                              hipStream_t stream) {
    const float* x      = (const float*)d_in[0];
    const float* ln1_g  = (const float*)d_in[1];
    const float* ln1_b  = (const float*)d_in[2];
    const float* w_qkv  = (const float*)d_in[3];
    const float* w_out  = (const float*)d_in[4];
    const float* b_out  = (const float*)d_in[5];
    const float* ln2_g  = (const float*)d_in[6];
    const float* ln2_b  = (const float*)d_in[7];
    const float* w_ff1  = (const float*)d_in[8];
    const float* b_ff1  = (const float*)d_in[9];
    const float* w_ff2  = (const float*)d_in[10];
    const float* b_ff2  = (const float*)d_in[11];
    const float* w_halt = (const float*)d_in[12];
    const float* b_halt = (const float*)d_in[13];

    char* ws = (char*)d_ws;
    unsigned short* wqkvT = (unsigned short*)(ws + OFF_WQKVT);
    unsigned short* woutT = (unsigned short*)(ws + OFF_WOUTT);
    unsigned short* wff1T = (unsigned short*)(ws + OFF_WFF1T);
    unsigned short* wff2T = (unsigned short*)(ws + OFF_WFF2T);
    unsigned short* hbuf  = (unsigned short*)(ws + OFF_A);   // LN1 out
    unsigned short* aout  = (unsigned short*)(ws + OFF_A);   // attention out (reuses h)
    unsigned short* h2    = (unsigned short*)(ws + OFF_A);   // LN2 out (reuses aout)
    unsigned short* qb    = (unsigned short*)(ws + OFF_Q);
    unsigned short* kbuf  = (unsigned short*)(ws + OFF_K);
    unsigned short* vtb   = (unsigned short*)(ws + OFF_V);
    unsigned short* gbuf  = (unsigned short*)(ws + OFF_G);   // reuses q/k/v
    float*          x1    = (float*)(ws + OFF_X1);

    float* xout = (float*)d_out;
    float* halt = xout + (size_t)BB * NSEQ * DD;

    // 1) weight convert+transpose
    wconv_kernel<<<(DD * 3 * INNER + 255) / 256, 256, 0, stream>>>(w_qkv, wqkvT, DD, 3 * INNER);
    wconv_kernel<<<(INNER * DD + 255) / 256, 256, 0, stream>>>(w_out, woutT, INNER, DD);
    wconv_kernel<<<(DD * FFD + 255) / 256, 256, 0, stream>>>(w_ff1, wff1T, DD, FFD);
    wconv_kernel<<<(FFD * DD + 255) / 256, 256, 0, stream>>>(w_ff2, wff2T, FFD, DD);
    // 2) LN1
    ln_kernel<<<MM, 256, 0, stream>>>(x, ln1_g, ln1_b, hbuf);
    // 3) QKV projection
    gemm_qkv_kernel<<<dim3(MM / 32, (3 * INNER) / 64), 32, 0, stream>>>(hbuf, wqkvT, qb, kbuf, vtb);
    // 4) causal attention (TDM-staged K stripes)
    attn_kernel<<<BB * HH * (NSEQ / 16), 32, 0, stream>>>(qb, kbuf, vtb, aout);
    // 5) out-proj + residual
    gemm_outproj_kernel<<<dim3(MM / 32, DD / 64), 32, 0, stream>>>(aout, woutT, b_out, x, x1);
    // 6) LN2
    ln_kernel<<<MM, 256, 0, stream>>>(x1, ln2_g, ln2_b, h2);
    // 7) FF1 + GELU
    gemm_ff1_kernel<<<dim3(MM / 32, FFD / 64), 32, 0, stream>>>(h2, wff1T, b_ff1, gbuf);
    // 8) FF2 + residual -> output x
    gemm_ff2_kernel<<<dim3(MM / 32, DD / 64), 32, 0, stream>>>(gbuf, wff2T, b_ff2, x1, xout);
    // 9) halt logits
    halt_kernel<<<BB, 256, 0, stream>>>(xout, w_halt, b_halt, halt);
}
